// AGCNetCLF_36197984371105
// MI455X (gfx1250) — matrix operand order repaired and checked by reading, not compiled, not run on hardware
//
// AGCNet forward for MI455X (gfx1250, wave32, WMMA).
// Bandwidth-bound workload (~20 GB traffic vs ~320 GFLOP): all dense GEMMs use
// v_wmma_f32_16x16x32_f16 with f16 weight arena + in-LDS fp32->f16 A staging,
// fused activations, gather/concat folded into GEMM staging, atomic segment
// softmax only for random-dst edges, direct loops for contiguous segments.
// Round-1 fixes: removed always-true N guards around WMMA (kills SALU branch
// dance between v_wmma), preload all 8 B fragments before the WMMA burst
// (clause-able b128 loads, overlapped waits), float4 fast path for A staging.
// Workspace requirement ~420 MB.

#include <hip/hip_runtime.h>
#include <hip/hip_bf16.h>

typedef __attribute__((ext_vector_type(16))) _Float16 v16h;
typedef __attribute__((ext_vector_type(8)))  float    v8f;
typedef _Float16 f16;

#define ACT_NONE 0
#define ACT_LEAKY 1
#define ACT_RELU 2
#define ACT_ELU 3
#define ACT_SIGMOID 4

__device__ __forceinline__ float actf(float x, int act) {
  switch (act) {
    case ACT_LEAKY:   return x > 0.f ? x : 0.01f * x;
    case ACT_RELU:    return x > 0.f ? x : 0.f;
    case ACT_ELU:     return x > 0.f ? x : (__expf(x) - 1.f);
    case ACT_SIGMOID: return 1.f / (1.f + __expf(-x));
    default:          return x;
  }
}

// ---------------- WMMA GEMM ----------------
// C[M,N] = act( concat_rows(A0[g0(r)], A1[g1(r)]) @ Wf16[N,Kpad]^T + bias )
// 256 threads = 8 waves; wave w owns rows [bx*128+w*16, +16), 8 n-tiles of 16.
// REQUIRES: each grid.y block covers a full 128 columns (all uses: N in {128,384}).
__global__ __launch_bounds__(256)
void agc_gemm(const float* __restrict__ A0, const int* __restrict__ g0, int lda0, int K0,
              const float* __restrict__ A1, const int* __restrict__ g1, int lda1, int K1,
              const f16* __restrict__ W, int Kpad, const float* __restrict__ bias,
              float* __restrict__ C, int ldc, int M, int N, int act) {
  __shared__ f16 sA[8][16 * 32];
  const int wave = threadIdx.x >> 5;
  const int lane = threadIdx.x & 31;
  const int row0 = blockIdx.x * 128 + wave * 16;
  const int ncol0 = blockIdx.y * 128;
  const int K = K0 + K1;

  v8f acc[8] = {};

  // staging assignment: lane -> (row sr, k-halfslab sk of 16 floats)
  const int sr = lane >> 1;
  const int sk = (lane & 1) * 16;
  int grow = row0 + sr;
  if (grow >= M) grow = M - 1;
  const int r0 = g0 ? g0[grow] : grow;
  const int r1 = g1 ? g1[grow] : grow;
  const float* a0p = A0 ? (A0 + (long)r0 * lda0) : nullptr;
  const float* a1p = A1 ? (A1 + (long)r1 * lda1) : nullptr;
  const bool al0 = a0p && ((lda0 & 3) == 0);                     // 16B-aligned rows
  const bool al1 = a1p && ((lda1 & 3) == 0) && ((K0 & 15) == 0); // slab alignment in A1

  const int ar  = lane & 15;           // A-frag row
  const int khi = (lane >> 4) * 8;     // A-frag k offset is 2*khi = 16h (contiguous k = 16h + s, same as B)
  const int bko = (lane >> 4) * 16;    // B-frag k offset (contiguous k = 16h + s, same as A)
  const int nloc = lane & 15;
  const f16* wrow = W + (long)(ncol0 + nloc) * Kpad + bko;

  for (int kb = 0; kb < Kpad; kb += 32) {
    // ---- stage 16x32 fp32 tile -> f16 LDS (wave-private region) ----
    f16* sp = &sA[wave][sr * 32 + sk];
    const int kslab = kb + sk;
    if (al0 && kslab + 16 <= K0) {
      const float4* p = (const float4*)(a0p + kslab);
      float4 q0 = p[0], q1 = p[1], q2 = p[2], q3 = p[3];
      sp[0] = (f16)q0.x; sp[1] = (f16)q0.y; sp[2] = (f16)q0.z; sp[3] = (f16)q0.w;
      sp[4] = (f16)q1.x; sp[5] = (f16)q1.y; sp[6] = (f16)q1.z; sp[7] = (f16)q1.w;
      sp[8] = (f16)q2.x; sp[9] = (f16)q2.y; sp[10] = (f16)q2.z; sp[11] = (f16)q2.w;
      sp[12] = (f16)q3.x; sp[13] = (f16)q3.y; sp[14] = (f16)q3.z; sp[15] = (f16)q3.w;
    } else if (al1 && kslab >= K0 && kslab + 16 <= K) {
      const float4* p = (const float4*)(a1p + (kslab - K0));
      float4 q0 = p[0], q1 = p[1], q2 = p[2], q3 = p[3];
      sp[0] = (f16)q0.x; sp[1] = (f16)q0.y; sp[2] = (f16)q0.z; sp[3] = (f16)q0.w;
      sp[4] = (f16)q1.x; sp[5] = (f16)q1.y; sp[6] = (f16)q1.z; sp[7] = (f16)q1.w;
      sp[8] = (f16)q2.x; sp[9] = (f16)q2.y; sp[10] = (f16)q2.z; sp[11] = (f16)q2.w;
      sp[12] = (f16)q3.x; sp[13] = (f16)q3.y; sp[14] = (f16)q3.z; sp[15] = (f16)q3.w;
    } else {
#pragma unroll
      for (int t = 0; t < 16; ++t) {
        int k = kslab + t;
        float v = 0.f;
        if (k < K0)      v = a0p[k];
        else if (k < K)  v = a1p[k - K0];
        sp[t] = (f16)v;
      }
    }
    if (a0p && kb + 32 < K0) __builtin_prefetch(a0p + kb + 32, 0, 1);
    __syncthreads();

    // ---- A fragment from LDS ----
    v16h af;
#pragma unroll
    for (int i = 0; i < 8; ++i) {
      af[i]     = sA[wave][ar * 32 + 2 * khi + i];
      af[i + 8] = sA[wave][ar * 32 + 2 * khi + 8 + i];
    }

    // ---- preload all 8 B fragments, then WMMA burst ----
    v16h bf[8];
#pragma unroll
    for (int t = 0; t < 8; ++t) {
      const f16* wp = wrow + (long)t * 16 * Kpad + kb;
#pragma unroll
      for (int i = 0; i < 16; ++i) bf[t][i] = wp[i];
    }
#pragma unroll
    for (int t = 0; t < 8; ++t) {
      acc[t] = __builtin_amdgcn_wmma_f32_16x16x32_f16(
          false, af, false, bf[t], (short)0, acc[t], false, false);
    }
    __syncthreads();
  }

  // ---- epilogue: bias + activation ----
  const int rbase = row0 + (lane >> 4) * 8;
#pragma unroll
  for (int t = 0; t < 8; ++t) {
    int n = ncol0 + t * 16 + nloc;
    float bv = bias ? bias[n] : 0.f;
#pragma unroll
    for (int i = 0; i < 8; ++i) {
      int r = rbase + i;
      if (r < M) C[(long)r * ldc + n] = actf(acc[t][i] + bv, act);
    }
  }
}

// ---------------- small kernels ----------------
__global__ void agc_convw(const float* __restrict__ W, f16* __restrict__ O,
                          int N, int K, int Kp) {
  int i = blockIdx.x * 256 + threadIdx.x;
  if (i >= N * Kp) return;
  int n = i / Kp, k = i - n * Kp;
  O[i] = (f16)(k < K ? W[(long)n * K + k] : 0.f);
}

__global__ void agc_fillf(float* p, float v, long n) {
  long i = (long)blockIdx.x * 256 + threadIdx.x;
  if (i < n) p[i] = v;
}
__global__ void agc_filli(int* p, int v, long n) {
  long i = (long)blockIdx.x * 256 + threadIdx.x;
  if (i < n) p[i] = v;
}

// N=1 linears: out[r] = act( dot(concat(A0[g0(r)],A1[g1(r)]), w) + b[0] )
__global__ void agc_rowdot(const float* __restrict__ A0, const int* __restrict__ g0,
                           int lda0, int K0,
                           const float* __restrict__ A1, const int* __restrict__ g1,
                           int lda1, int K1,
                           const float* __restrict__ w, const float* __restrict__ b,
                           float* __restrict__ out, int M, int act) {
  int r = blockIdx.x * 256 + threadIdx.x;
  if (r >= M) return;
  float s = b[0];
  const float* p0 = A0 + (long)(g0 ? g0[r] : r) * lda0;
#pragma unroll 4
  for (int k = 0; k < K0; ++k) s += p0[k] * w[k];
  if (A1) {
    const float* p1 = A1 + (long)(g1 ? g1[r] : r) * lda1;
#pragma unroll 4
    for (int k = 0; k < K1; ++k) s += p1[k] * w[K0 + k];
  }
  out[r] = actf(s, act);
}

__device__ __forceinline__ int f2key(float f) {
  int i = __float_as_int(f);
  return i >= 0 ? i : (i ^ 0x7fffffff);
}
__device__ __forceinline__ float key2f(int k) {
  return __int_as_float(k >= 0 ? k : (k ^ 0x7fffffff));
}

__global__ void agc_segamax(const float* __restrict__ l, const int* __restrict__ seg,
                            int* __restrict__ smax, int E) {
  int e = blockIdx.x * 256 + threadIdx.x;
  if (e >= E) return;
  atomicMax(&smax[seg[e]], f2key(l[e]));
}
__global__ void agc_expsum(const float* __restrict__ l, const int* __restrict__ seg,
                           const int* __restrict__ smax, float* __restrict__ ex,
                           float* __restrict__ ssum, int E) {
  int e = blockIdx.x * 256 + threadIdx.x;
  if (e >= E) return;
  int s = seg[e];
  float v = __expf(l[e] - key2f(smax[s]));
  ex[e] = v;
  atomicAdd(&ssum[s], v);
}
// out[seg[e],:] += (ex[e]/ssum[seg[e]]) * V[e,:]
__global__ void agc_scatter(const float* __restrict__ ex, const float* __restrict__ ssum,
                            const int* __restrict__ seg, const float* __restrict__ V,
                            float* __restrict__ out, int E) {
  long i = (long)blockIdx.x * 256 + threadIdx.x;
  if (i >= (long)E * 128) return;
  int e = (int)(i >> 7), d = (int)(i & 127);
  int s = seg[e];
  float a = ex[e] / ssum[s];
  atomicAdd(&out[(long)s * 128 + d], a * V[(long)e * 128 + d]);
}

__global__ void agc_act_ip(float* x, long n, int act) {
  long i = (long)blockIdx.x * 256 + threadIdx.x;
  if (i < n) x[i] = actf(x[i], act);
}

// contiguous segments of fixed length
__global__ void agc_segsum(const float* __restrict__ in, float* __restrict__ out,
                           int nseg, int len, int act) {
  long i = (long)blockIdx.x * 256 + threadIdx.x;
  if (i >= (long)nseg * 128) return;
  int s = (int)(i >> 7), d = (int)(i & 127);
  const float* p = in + (long)s * len * 128 + d;
  float acc = 0.f;
  for (int j = 0; j < len; ++j) acc += p[(long)j * 128];
  out[(long)s * 128 + d] = actf(acc, act);
}
__global__ void agc_segsum_w(const float* __restrict__ in, const float* __restrict__ a,
                             float* __restrict__ out, int nseg, int len, int act) {
  long i = (long)blockIdx.x * 256 + threadIdx.x;
  if (i >= (long)nseg * 128) return;
  int s = (int)(i >> 7), d = (int)(i & 127);
  const float* p = in + (long)s * len * 128 + d;
  const float* aw = a + (long)s * len;
  float acc = 0.f;
  for (int j = 0; j < len; ++j) acc += aw[j] * p[(long)j * 128];
  out[(long)s * 128 + d] = actf(acc, act);
}
// softmax over contiguous segments (one thread per segment, len <= 12)
__global__ void agc_softseg(const float* __restrict__ l, float* __restrict__ a,
                            int nseg, int len) {
  int s = blockIdx.x * 256 + threadIdx.x;
  if (s >= nseg) return;
  const float* p = l + (long)s * len;
  float* q = a + (long)s * len;
  float m = -3.4e38f;
  for (int j = 0; j < len; ++j) m = fmaxf(m, p[j]);
  float sum = 0.f;
  for (int j = 0; j < len; ++j) { float v = __expf(p[j] - m); q[j] = v; sum += v; }
  float inv = 1.f / sum;
  for (int j = 0; j < len; ++j) q[j] *= inv;
}

// torch GRUCell: r,z,n gate order
__global__ void agc_gru(const float* __restrict__ gi, const float* __restrict__ gh,
                        const float* __restrict__ h, int ldh,
                        float* __restrict__ out, int ldo, int n) {
  long i = (long)blockIdx.x * 256 + threadIdx.x;
  if (i >= (long)n * 128) return;
  int r = (int)(i >> 7), d = (int)(i & 127);
  const float* pi = gi + (long)r * 384;
  const float* ph = gh + (long)r * 384;
  float rr = 1.f / (1.f + __expf(-(pi[d] + ph[d])));
  float zz = 1.f / (1.f + __expf(-(pi[128 + d] + ph[128 + d])));
  float nn = tanhf(pi[256 + d] + rr * ph[256 + d]);
  out[(long)r * ldo + d] = (1.f - zz) * nn + zz * h[(long)r * ldh + d];
}

__global__ void agc_copy(const float* __restrict__ src, int lds, float* __restrict__ dst,
                         int ldd, int rows, int cols) {
  long i = (long)blockIdx.x * 256 + threadIdx.x;
  if (i >= (long)rows * cols) return;
  int r = (int)(i / cols), c = (int)(i - (long)r * cols);
  dst[(long)r * ldd + c] = src[(long)r * lds + c];
}
__global__ void agc_addinto(const float* __restrict__ src, float* __restrict__ dst, long n) {
  long i = (long)blockIdx.x * 256 + threadIdx.x;
  if (i < n) dst[i] += src[i];
}
__global__ void agc_scale_relu(const float* __restrict__ src, float* __restrict__ dst,
                               float sc, long n) {
  long i = (long)blockIdx.x * 256 + threadIdx.x;
  if (i < n) dst[i] = fmaxf(src[i] * sc, 0.f);
}

__global__ void agc_bnstats(const float* __restrict__ x, int ld, int M,
                            float* __restrict__ mean, float* __restrict__ rstd) {
  int c = blockIdx.x * 256 + threadIdx.x;
  if (c >= 128) return;
  float s = 0.f, s2 = 0.f;
  for (int r = 0; r < M; ++r) { float v = x[(long)r * ld + c]; s += v; s2 += v * v; }
  float m = s / M;
  mean[c] = m;
  rstd[c] = rsqrtf(s2 / M - m * m + 1e-5f);
}
__global__ void agc_bnapply(float* __restrict__ x, int ld, int M,
                            const float* __restrict__ mean, const float* __restrict__ rstd,
                            const float* __restrict__ g, const float* __restrict__ b) {
  long i = (long)blockIdx.x * 256 + threadIdx.x;
  if (i >= (long)M * 128) return;
  int r = (int)(i >> 7), c = (int)(i & 127);
  float v = (x[(long)r * ld + c] - mean[c]) * rstd[c] * g[c] + b[c];
  x[(long)r * ld + c] = fmaxf(v, 0.f);
}

__global__ void agc_tbtc(const float* tb1, const float* tc1, const float* tb2,
                         const float* tc2, float* o, int n) {
  int r = blockIdx.x * 256 + threadIdx.x;
  if (r >= n) return;
  o[r * 4 + 0] = tb1[r]; o[r * 4 + 1] = tc1[r];
  o[r * 4 + 2] = tb2[r]; o[r * 4 + 3] = tc2[r];
}

// ---------------- host side ----------------
struct LinP { const float* W; const float* b; };
struct GruP { const float *Wh, *Wi, *bh, *bi; };
struct AtomP { LinP att[2]; LinP ctx[2]; GruP gru[2]; LinP ne; };
struct MolP { LinP att, ctx; GruP gru; };
struct CompP {
  LinP emb_fe, emb_fn, emb_me, emb_mn, fatt;
  AtomP fa[4]; MolP fm[4];
  AtomP ja[4]; MolP jm[4]; LinP proj[4];
};
struct BnP { const float *b, *g; };
struct LW { const f16* W; int Kp; const float* b; };
struct GW { LW ih, hh; };
struct AtomW { LW ne; LW ctx[2]; GW gru[2]; };
struct MolW { LW ctx; GW gru; };
struct CompW { LW emb_fe, emb_fn, emb_me, emb_mn, fatt; AtomW fa[4]; MolW fm[4];
               AtomW ja[4]; MolW jm[4]; LW proj[4]; };
struct CompIn {
  const float *fnode, *fedge, *mnode, *medge, *tb, *tc;
  const int *fsrc, *fdst, *fseg, *msrc, *mdst, *mseg;
};

#define LT(kern, total, ...)                                                     \
  kern<<<dim3((unsigned)((((long)(total)) + 255) / 256)), dim3(256), 0, stream>>>(__VA_ARGS__)

extern "C" void kernel_launch(void* const* d_in, const int* in_sizes, int n_in,
                              void* d_out, int out_size, void* d_ws, size_t ws_size,
                              hipStream_t stream) {
  (void)in_sizes; (void)n_in; (void)out_size;
  const int NMOL = 512, NMOTIF = 4096, EMOTIF = 8192, NATOM = 49152, EATOM = 98304;
  const int FLEN = NATOM / NMOTIF;   // 12 atoms per motif (contiguous fseg)
  const int MLEN = NMOTIF / NMOL;    // 8 motifs per mol (contiguous mseg)

  auto F = [&](int i) { return (const float*)d_in[i]; };
  auto I = [&](int i) { return (const int*)d_in[i]; };
  CompIn in1{F(0), F(1), F(5), F(6), F(10), F(11), I(2), I(3), I(4), I(7), I(8), I(9)};
  CompIn in2{F(12), F(13), F(17), F(18), F(22), F(23), I(14), I(15), I(16), I(19), I(20), I(21)};

  // ---- params: jax tree_flatten order (dict keys sorted), leaves from d_in[24] ----
  int pi = 24;
  auto L = [&]() { LinP l{F(pi), F(pi + 1)}; pi += 2; return l; };
  auto G = [&]() { GruP g{F(pi), F(pi + 1), F(pi + 2), F(pi + 3)}; pi += 4; return g; }; // Wh,Wi,bh,bi
  auto AP = [&]() { AtomP a; a.att[0] = L(); a.att[1] = L(); a.ctx[0] = L(); a.ctx[1] = L();
                    a.gru[0] = G(); a.gru[1] = G(); a.ne = L(); return a; };              // att,ctx,gru,ne
  auto MP = [&]() { MolP m; m.att = L(); m.ctx = L(); m.gru = G(); return m; };          // att,ctx,gru
  auto CP = [&]() { CompP c;
    c.emb_fe = L(); c.emb_fn = L(); c.emb_me = L(); c.emb_mn = L(); c.fatt = L();
    for (int h = 0; h < 4; ++h) { c.fa[h] = AP(); c.fm[h] = MP(); }
    for (int h = 0; h < 4; ++h) { c.ja[h] = AP(); c.jm[h] = MP(); c.proj[h] = L(); }
    return c; };
  CompP C1 = CP(), C2 = CP();
  BnP bn11{F(pi), F(pi + 1)}; pi += 2; LinP p11 = L();
  BnP bn12{F(pi), F(pi + 1)}; pi += 2; LinP p12 = L();
  BnP bn13{F(pi), F(pi + 1)}; pi += 2; LinP p13 = L();
  LinP p3 = L(); LinP p4 = L();

  // ---- workspace bump allocator ----
  char* wsb = (char*)d_ws;
  size_t off = 0; (void)ws_size;
  auto alloc = [&](size_t bytes) { size_t o = (off + 255) & ~(size_t)255; off = o + bytes; return (void*)(wsb + o); };
  auto aF = [&](size_t n) { return (float*)alloc(n * sizeof(float)); };
  auto aI = [&](size_t n) { return (int*)alloc(n * sizeof(int)); };

  float* fnA   = aF((size_t)NATOM * 128);
  float* nodeB = aF((size_t)NATOM * 128);
  float* feB   = aF((size_t)EATOM * 128);
  float* nb0   = aF((size_t)EATOM * 128);
  float* Vb    = aF((size_t)EATOM * 128);
  float* gib   = aF((size_t)NATOM * 384);
  float* ghb   = aF((size_t)NATOM * 384);
  float* ctxb  = aF((size_t)NATOM * 128);
  float* elog  = aF(EATOM);
  float* exb   = aF(EATOM);
  int*   smax  = aI(NATOM);
  float* ssum  = aF(NATOM);
  float* nodew = aF(NATOM);
  float* supM  = aF((size_t)NMOTIF * 128);
  float* mctx  = aF((size_t)NMOTIF * 128);
  float* fragcat = aF((size_t)NMOTIF * 512);
  float* gmotif  = aF((size_t)NMOTIF * 128);
  float* mnEmb   = aF((size_t)NMOTIF * 128);
  float* meB     = aF((size_t)EMOTIF * 128);
  float* gacc  = aF((size_t)NMOL * 128);
  float* sup1  = aF((size_t)NMOL * 128);
  float* sup2  = aF((size_t)NMOL * 128);
  float* tbtc  = aF((size_t)NMOL * 4);
  float* feat  = aF((size_t)NMOL * 384);
  float* dbuf  = aF((size_t)NMOL * 128);
  float* bnm = aF(128); float* bnr = aF(128);

  // ---- convert all GEMM weights to f16 (padded K) ----
  auto conv = [&](const LinP& l, int N, int K) -> LW {
    int Kp = (K + 31) & ~31;
    f16* dst = (f16*)alloc((size_t)N * Kp * sizeof(f16));
    LT(agc_convw, N * Kp, l.W, dst, N, K, Kp);
    return LW{dst, Kp, l.b};
  };
  auto convG = [&](const GruP& g) -> GW {
    LinP wi{g.Wi, g.bi}, wh{g.Wh, g.bh};
    GW o; o.ih = conv(wi, 384, 128); o.hh = conv(wh, 384, 128); return o;
  };
  auto convA = [&](const AtomP& a) -> AtomW {
    AtomW o; o.ne = conv(a.ne, 128, 256);
    o.ctx[0] = conv(a.ctx[0], 128, 128); o.ctx[1] = conv(a.ctx[1], 128, 128);
    o.gru[0] = convG(a.gru[0]); o.gru[1] = convG(a.gru[1]); return o;
  };
  auto convM = [&](const MolP& m) -> MolW {
    MolW o; o.ctx = conv(m.ctx, 128, 128); o.gru = convG(m.gru); return o;
  };
  auto convC = [&](const CompP& c) -> CompW {
    CompW o;
    o.emb_fe = conv(c.emb_fe, 128, 10); o.emb_fn = conv(c.emb_fn, 128, 39);
    o.emb_me = conv(c.emb_me, 128, 10); o.emb_mn = conv(c.emb_mn, 128, 82);
    o.fatt = conv(c.fatt, 128, 512);
    for (int h = 0; h < 4; ++h) { o.fa[h] = convA(c.fa[h]); o.fm[h] = convM(c.fm[h]); }
    for (int h = 0; h < 4; ++h) { o.ja[h] = convA(c.ja[h]); o.jm[h] = convM(c.jm[h]);
                                  o.proj[h] = conv(c.proj[h], 128, 256); }
    return o;
  };
  CompW W1 = convC(C1), W2 = convC(C2);
  LW p11w = conv(p11, 128, 128), p12w = conv(p12, 128, 128);
  LW p13w = conv(p13, 128, 4),  p3w = conv(p3, 128, 384);

  // ---- launchers ----
  auto gemm = [&](const float* A0, const int* g0, int lda0, int K0,
                  const float* A1, const int* g1, int lda1, int K1,
                  const LW& w, float* Cp, int ldc, int M, int N, int act) {
    dim3 grid((M + 127) / 128, (N + 127) / 128);
    agc_gemm<<<grid, dim3(256), 0, stream>>>(A0, g0, lda0, K0, A1, g1, lda1, K1,
                                             w.W, w.Kp, w.b, Cp, ldc, M, N, act);
  };

  auto atom_fp = [&](const AtomP& ap, const AtomW& aw, float* node, const float* ef,
                     const int* src, const int* dst, int n, int e) {
    // nb0 = leaky(lin(concat(node[src], ef), ne))
    gemm(node, src, 128, 128, ef, nullptr, 128, 128, aw.ne, nb0, 128, e, 128, ACT_LEAKY);
    for (int l = 0; l < 2; ++l) {
      const float* hnb = (l == 0) ? nb0 : node;
      const int* gnb = (l == 0) ? nullptr : src;
      // attention logits, leaky
      LT(agc_rowdot, e, node, dst, 128, 128, hnb, gnb, 128, 128,
         ap.att[l].W, ap.att[l].b, elog, e, ACT_LEAKY);
      // segment softmax over random dst (atomics)
      LT(agc_filli, n, smax, (int)0x80000000, (long)n);
      LT(agc_fillf, n, ssum, 0.f, (long)n);
      LT(agc_segamax, e, elog, dst, smax, e);
      LT(agc_expsum, e, elog, dst, smax, exb, ssum, e);
      // V = lin(h_nb, ctx)
      gemm(hnb, gnb, 128, 128, nullptr, nullptr, 0, 0, aw.ctx[l], Vb, 128, e, 128, ACT_NONE);
      LT(agc_fillf, (long)n * 128, ctxb, 0.f, (long)n * 128);
      LT(agc_scatter, (long)e * 128, exb, ssum, dst, Vb, ctxb, e);
      LT(agc_act_ip, (long)n * 128, ctxb, (long)n * 128, ACT_ELU);
      // GRU gates + fuse
      gemm(ctxb, nullptr, 128, 128, nullptr, nullptr, 0, 0, aw.gru[l].ih, gib, 384, n, 384, ACT_NONE);
      gemm(node, nullptr, 128, 128, nullptr, nullptr, 0, 0, aw.gru[l].hh, ghb, 384, n, 384, ACT_NONE);
      LT(agc_gru, (long)n * 128, gib, ghb, node, 128, node, 128, n);
    }
  };

  auto mol_fp = [&](const MolP& mp, const MolW& mw, const float* node, const int* seg,
                    int n, int len, int ng, float* sup) {
    LT(agc_segsum, (long)ng * 128, node, sup, ng, len, ACT_NONE);
    for (int t = 0; t < 2; ++t) {
      LT(agc_rowdot, n, sup, seg, 128, 128, node, nullptr, 128, 128,
         mp.att.W, mp.att.b, elog, n, ACT_LEAKY);
      LT(agc_softseg, ng, elog, nodew, ng, len);
      gemm(node, nullptr, 128, 128, nullptr, nullptr, 0, 0, mw.ctx, Vb, 128, n, 128, ACT_NONE);
      LT(agc_segsum_w, (long)ng * 128, Vb, nodew, mctx, ng, len, ACT_ELU);
      gemm(mctx, nullptr, 128, 128, nullptr, nullptr, 0, 0, mw.gru.ih, gib, 384, ng, 384, ACT_NONE);
      gemm(sup, nullptr, 128, 128, nullptr, nullptr, 0, 0, mw.gru.hh, ghb, 384, ng, 384, ACT_NONE);
      LT(agc_gru, (long)ng * 128, gib, ghb, sup, 128, sup, 128, ng);
    }
  };

  auto component = [&](const CompP& cp, const CompW& cw, const CompIn& ci, float* outsup) {
    gemm(ci.fnode, nullptr, 39, 39, nullptr, nullptr, 0, 0, cw.emb_fn, fnA, 128, NATOM, 128, ACT_LEAKY);
    gemm(ci.fedge, nullptr, 10, 10, nullptr, nullptr, 0, 0, cw.emb_fe, feB, 128, EATOM, 128, ACT_LEAKY);
    for (int h = 0; h < 4; ++h) {
      LT(agc_copy, (long)NATOM * 128, fnA, 128, nodeB, 128, NATOM, 128);
      atom_fp(cp.fa[h], cw.fa[h], nodeB, feB, ci.fsrc, ci.fdst, NATOM, EATOM);
      mol_fp(cp.fm[h], cw.fm[h], nodeB, ci.fseg, NATOM, FLEN, NMOTIF, supM);
      LT(agc_copy, (long)NMOTIF * 128, supM, 128, fragcat + h * 128, 512, NMOTIF, 128);
    }
    gemm(fragcat, nullptr, 512, 512, nullptr, nullptr, 0, 0, cw.fatt, gmotif, 128, NMOTIF, 128, ACT_RELU);
    gemm(ci.mnode, nullptr, 82, 82, nullptr, nullptr, 0, 0, cw.emb_mn, mnEmb, 128, NMOTIF, 128, ACT_LEAKY);
    gemm(ci.medge, nullptr, 10, 10, nullptr, nullptr, 0, 0, cw.emb_me, meB, 128, EMOTIF, 128, ACT_LEAKY);
    LT(agc_fillf, (long)NMOL * 128, gacc, 0.f, (long)NMOL * 128);
    for (int h = 0; h < 4; ++h) {
      // x = lin(concat(graph_motif, mn_emb), proj)   (no activation)
      gemm(gmotif, nullptr, 128, 128, mnEmb, nullptr, 128, 128, cw.proj[h],
           nodeB, 128, NMOTIF, 128, ACT_NONE);
      atom_fp(cp.ja[h], cw.ja[h], nodeB, meB, ci.msrc, ci.mdst, NMOTIF, EMOTIF);
      mol_fp(cp.jm[h], cw.jm[h], nodeB, ci.mseg, NMOTIF, MLEN, NMOL, supM);
      LT(agc_addinto, (long)NMOL * 128, supM, gacc, (long)NMOL * 128);
    }
    LT(agc_scale_relu, (long)NMOL * 128, gacc, outsup, 0.25f, (long)NMOL * 128);
  };

  component(C1, W1, in1, sup1);
  component(C2, W2, in2, sup2);

  // ---- head MLP: d1|d2|d3 written directly into feat[512,384] ----
  gemm(sup1, nullptr, 128, 128, nullptr, nullptr, 0, 0, p11w, feat + 0, 384, NMOL, 128, ACT_NONE);
  LT(agc_bnstats, 128, feat + 0, 384, NMOL, bnm, bnr);
  LT(agc_bnapply, (long)NMOL * 128, feat + 0, 384, NMOL, bnm, bnr, bn11.g, bn11.b);

  gemm(sup2, nullptr, 128, 128, nullptr, nullptr, 0, 0, p12w, feat + 128, 384, NMOL, 128, ACT_NONE);
  LT(agc_bnstats, 128, feat + 128, 384, NMOL, bnm, bnr);
  LT(agc_bnapply, (long)NMOL * 128, feat + 128, 384, NMOL, bnm, bnr, bn12.g, bn12.b);

  LT(agc_tbtc, NMOL, in1.tb, in1.tc, in2.tb, in2.tc, tbtc, NMOL);
  gemm(tbtc, nullptr, 4, 4, nullptr, nullptr, 0, 0, p13w, feat + 256, 384, NMOL, 128, ACT_NONE);
  LT(agc_bnstats, 128, feat + 256, 384, NMOL, bnm, bnr);
  LT(agc_bnapply, (long)NMOL * 128, feat + 256, 384, NMOL, bnm, bnr, bn13.g, bn13.b);

  gemm(feat, nullptr, 384, 384, nullptr, nullptr, 0, 0, p3w, dbuf, 128, NMOL, 128, ACT_RELU);
  LT(agc_rowdot, NMOL, dbuf, nullptr, 128, 128, nullptr, nullptr, 0, 0,
     p4.W, p4.b, (float*)d_out, NMOL, ACT_SIGMOID);
}
